// M2DBlock_23785528885961
// MI455X (gfx1250) — matrix operand order repaired and checked by reading, not compiled, not run on hardware
//
#include <hip/hip_runtime.h>
#include <math.h>

// ---------------------------------------------------------------------------
// Problem constants (from the reference)
// ---------------------------------------------------------------------------
#define BB      2
#define HH      64
#define WW      64
#define DMODEL  192
#define DINNER  192
#define NSTATE  16
#define RRANK   12
#define MTOK    (BB * HH * WW)        // 8192 tokens
#define PROJ_LD 80                    // 72 proj channels padded to 80
#define NPROJ   (2 * RRANK + 3 * NSTATE)   // 72

typedef __attribute__((ext_vector_type(16))) __bf16 v16bf;
typedef __attribute__((ext_vector_type(8)))  __bf16 v8bf;
typedef __attribute__((ext_vector_type(4)))  __bf16 v4bf;
typedef __attribute__((ext_vector_type(8)))  float  v8f;

__device__ __forceinline__ float gelu_f(float x) {
    // exact GELU (approximate=False): 0.5*x*(1+erf(x/sqrt(2)))
    return 0.5f * x * (1.0f + erff(x * 0.70710678118654752440f));
}

__device__ __forceinline__ float softplus_f(float x) {
    return (x > 20.0f) ? x : log1pf(__expf(x));
}

// ---------------------------------------------------------------------------
// One-shot f32 -> bf16 conversion (4-wide). Weights are converted exactly
// once instead of being re-converted by all 512 M-blocks of every GEMM.
// ---------------------------------------------------------------------------
__global__ __launch_bounds__(256) void cvt_bf16_kernel(
    const float* __restrict__ src, __bf16* __restrict__ dst, int count4)
{
    const int i = blockIdx.x * 256 + threadIdx.x;
    if (i >= count4) return;
    const float4 v = *(const float4*)(src + 4 * (size_t)i);
    v4bf o;
    o[0] = (__bf16)v.x; o[1] = (__bf16)v.y; o[2] = (__bf16)v.z; o[3] = (__bf16)v.w;
    *(v4bf*)(dst + 4 * (size_t)i) = o;
}

// ---------------------------------------------------------------------------
// bf16 WMMA GEMM:  Out[M,N] = act( X[M,K] * W[N,K]^T + bias[N] )
//
// A and B are pre-converted bf16, 32B-aligned rows -> the hot loop is pure
// global_load_b128 + v_wmma (no cvt on the critical path).
// KC compile-time -> K-loop fully unrolls (KC/32 WMMA steps); each wave
// computes NTPW adjacent 16x16 tiles, reusing the A fragment NTPW times.
//
// Fragment layouts (CDNA5 v_wmma_f32_16x16x32_bf16, wave32):
//   A (16x32, MxK): lane l -> row m0+(l&15); halves e hold K = k0+16*(l>=16)+e
//   B (32x16, KxN): lane l -> col n0+(l&15); halves e hold K = k0+16*(l>=16)+e
//   C/D (16x16):    vgpr r -> row m0 + r + 8*(l>=16), col n0+(l&15)
//
// Ragged N: B column n lives entirely in lanes {n&15, (n&15)+16} and only
// contributes to output column n, which only those lanes would store; for
// n >= Nreal redirect the B pointer to row 0 and skip the store.
// ---------------------------------------------------------------------------
template <int KC, int NTPW>
__global__ __launch_bounds__(128) void gemm_bf16_wmma(
    const __bf16* __restrict__ X,     // M x KC      (bf16)
    const __bf16* __restrict__ W,     // Nreal x KC  (bf16)
    const float*  __restrict__ bias,  // Nreal or nullptr
    float*        __restrict__ Out,   // M x ldo     (f32, may be nullptr)
    __bf16*       __restrict__ Outb,  // M x ldo     (bf16 copy, may be nullptr)
    int Nreal, int ldo, int fuse_gelu)
{
    const int lane = threadIdx.x & 31;
    const int wave = threadIdx.x >> 5;
    const int m0   = blockIdx.x * 16;

    const int row  = m0 + (lane & 15);
    const int aoff = (lane >> 4) << 4;   // 0 or 16
    const int boff = (lane >> 4) << 4;   // 0 or 16
    const __bf16* pa_base = X + (size_t)row * KC + aoff;

    int           ncol[NTPW];
    bool          nval[NTPW];
    const __bf16* pb_base[NTPW];
    #pragma unroll
    for (int t = 0; t < NTPW; ++t) {
        ncol[t]    = (wave * NTPW + t) * 16 + (lane & 15);
        nval[t]    = (ncol[t] < Nreal);
        pb_base[t] = W + (size_t)(nval[t] ? ncol[t] : 0) * KC + boff;
    }

    v8f acc[NTPW] = {};

    #pragma unroll
    for (int k0 = 0; k0 < KC; k0 += 32) {
        const __bf16* pa = pa_base + k0;
        const v8bf alo = *(const v8bf*)pa;          // K = kb + 0..7
        const v8bf ahi = *(const v8bf*)(pa + 8);    // K = kb + 8..15
        const v16bf a = __builtin_shufflevector(alo, ahi,
                            0, 1, 2, 3, 4, 5, 6, 7, 8, 9, 10, 11, 12, 13, 14, 15);
        if (k0 + 32 < KC) {
            __builtin_prefetch((const void*)(pa_base + k0 + 32), 0, 0); // global_prefetch_b8
        }
        #pragma unroll
        for (int t = 0; t < NTPW; ++t) {
            const v16bf b = *(const v16bf*)(pb_base[t] + k0);  // 32B aligned
            acc[t] = __builtin_amdgcn_wmma_f32_16x16x32_bf16(
                         false, a, false, b, (short)0, acc[t], false, false);
        }
    }

    const int orow0 = m0 + ((lane >> 4) << 3);
    #pragma unroll
    for (int t = 0; t < NTPW; ++t) {
        if (nval[t]) {
            const float bval = bias ? bias[ncol[t]] : 0.0f;
            #pragma unroll
            for (int r = 0; r < 8; ++r) {
                float v = acc[t][r] + bval;
                if (fuse_gelu) v = gelu_f(v);
                const size_t o = (size_t)(orow0 + r) * ldo + ncol[t];
                if (Out)  Out[o]  = v;
                if (Outb) Outb[o] = (__bf16)v;
            }
        }
    }
}

// ---------------------------------------------------------------------------
// deltaT/deltaL = softplus(dt_r @ dt_proj_w^T + b)   (K = DT_RANK = 12)
// K is too small for WMMA to pay off; pure VALU, one thread per (token, d).
// ---------------------------------------------------------------------------
__global__ __launch_bounds__(256) void delta_kernel(
    const float* __restrict__ proj,      // M x PROJ_LD; [0:12]=dtT_r [12:24]=dtL_r
    const float* __restrict__ wT,        // DINNER x RRANK
    const float* __restrict__ bT,
    const float* __restrict__ wL,
    const float* __restrict__ bL,
    float* __restrict__ deltaT,
    float* __restrict__ deltaL,
    int total)
{
    const int idx = blockIdx.x * 256 + threadIdx.x;
    if (idx >= total) return;
    const int m = idx / DINNER;
    const int d = idx - m * DINNER;
    const float* r = proj + (size_t)m * PROJ_LD;
    float sT = bT[d], sL = bL[d];
    #pragma unroll
    for (int k = 0; k < RRANK; ++k) {
        sT += r[k]         * wT[d * RRANK + k];
        sL += r[RRANK + k] * wL[d * RRANK + k];
    }
    deltaT[idx] = softplus_f(sT);
    deltaL[idx] = softplus_f(sL);
}

// ---------------------------------------------------------------------------
// 2D wavefront SSM:  h[i,j] = dB + aT*h[i-1,j] + aL*h[i,j-1]
// One 1024-thread block per (b,d): thread (j = tid>>4, n = tid&15) owns the
// state cell for column j, state n, carrying h across rows in a register.
// Per row: the column recurrence is a first-order linear scan -> Kogge-Stone
// over 64 columns (6 steps) on the affine-map monoid (a,t).
// y[m,d] = gelu( sum_n h*C[m,n] + u*D_skip[d] ), emitted in bf16 (out_proj A).
// ---------------------------------------------------------------------------
__global__ __launch_bounds__(1024) void scan_kernel(
    const float* __restrict__ u,        // M x DINNER
    const float* __restrict__ proj,     // M x PROJ_LD
    const float* __restrict__ deltaT,   // M x DINNER
    const float* __restrict__ deltaL,   // M x DINNER
    const float* __restrict__ AT_log,   // DINNER x NSTATE
    const float* __restrict__ AL_log,   // DINNER x NSTATE
    const float* __restrict__ Dsk,      // DINNER
    __bf16* __restrict__ y)             // M x DINNER (bf16)
{
    __shared__ float sA[1024];
    __shared__ float sT[1024];

    const int tid = threadIdx.x;
    const int j   = tid >> 4;
    const int n   = tid & 15;
    const int b   = blockIdx.x / DINNER;
    const int d   = blockIdx.x - b * DINNER;

    const float AT  = -__expf(AT_log[d * NSTATE + n]);
    const float AL  = -__expf(AL_log[d * NSTATE + n]);
    const float dsk = Dsk[d];

    float h = 0.0f;                       // h[i-1][j][n]
    for (int i = 0; i < HH; ++i) {
        const int   m  = (b * HH + i) * WW + j;
        const float dT = deltaT[(size_t)m * DINNER + d];
        const float dL = deltaL[(size_t)m * DINNER + d];
        const float uu = u[(size_t)m * DINNER + d];
        const float* pr = proj + (size_t)m * PROJ_LD;
        const float bT = pr[24 + n];      // B_T
        const float bL = pr[40 + n];      // B_L
        const float cC = pr[56 + n];      // C

        const float aT = __expf(dT * AT);
        const float aL = __expf(dL * AL);
        const float dB = (dT * bT + dL * bL) * uu;

        // row input: top = dB + aT * h[i-1][j]; scan over j with gain aL
        sA[tid] = aL;
        sT[tid] = dB + aT * h;
        __syncthreads();
        #pragma unroll
        for (int off = 1; off < WW; off <<= 1) {
            float pa = 0.0f, pt = 0.0f;
            const bool has = (j >= off);
            if (has) { pa = sA[tid - (off << 4)]; pt = sT[tid - (off << 4)]; }
            __syncthreads();
            if (has) {
                const float ca = sA[tid];
                sA[tid] = ca * pa;
                sT[tid] = sT[tid] + ca * pt;
            }
            __syncthreads();
        }
        h = sT[tid];                      // inclusive scan result = h[i][j][n]

        // reduce over n: y[m,d] = gelu( sum_n h*C + u*Dskip )
        sT[tid] = h * cC;                 // own-cell overwrite, no race
        __syncthreads();
        if (n == 0) {
            float s = 0.0f;
            #pragma unroll
            for (int q = 0; q < NSTATE; ++q) s += sT[(j << 4) + q];
            y[(size_t)m * DINNER + d] = (__bf16)gelu_f(s + uu * dsk);
        }
        __syncthreads();
    }
}

// ---------------------------------------------------------------------------
// Launch: cvt(x,W) -> in_proj(WMMA+GELU, f32+bf16 u) -> x_proj(WMMA) -> delta
//         -> 2D scan (bf16 y) -> out_proj(WMMA)
// Workspace: f32 { u | proj(ld80) | deltaT | deltaL } then bf16 { xb | ub |
// yb | w_in | w_xp | w_out }  (~31 MB, all L2-resident)
// ---------------------------------------------------------------------------
extern "C" void kernel_launch(void* const* d_in, const int* in_sizes, int n_in,
                              void* d_out, int out_size, void* d_ws, size_t ws_size,
                              hipStream_t stream)
{
    const float* x      = (const float*)d_in[0];
    const float* w_in   = (const float*)d_in[1];
    const float* b_in   = (const float*)d_in[2];
    const float* w_xp   = (const float*)d_in[3];
    const float* w_dtT  = (const float*)d_in[4];
    const float* b_dtT  = (const float*)d_in[5];
    const float* w_dtL  = (const float*)d_in[6];
    const float* b_dtL  = (const float*)d_in[7];
    const float* AT_log = (const float*)d_in[8];
    const float* AL_log = (const float*)d_in[9];
    const float* Dsk    = (const float*)d_in[10];
    const float* w_out  = (const float*)d_in[11];
    const float* b_out  = (const float*)d_in[12];
    float*       out    = (float*)d_out;

    // f32 region
    float* u     = (float*)d_ws;
    float* proj  = u    + (size_t)MTOK * DINNER;
    float* delT  = proj + (size_t)MTOK * PROJ_LD;
    float* delL  = delT + (size_t)MTOK * DINNER;
    // bf16 region (starts 16B-aligned; all sub-array sizes are multiples of 8)
    __bf16* xb     = (__bf16*)(delL + (size_t)MTOK * DINNER);
    __bf16* ub     = xb     + (size_t)MTOK * DMODEL;
    __bf16* yb     = ub     + (size_t)MTOK * DINNER;
    __bf16* w_in_b = yb     + (size_t)MTOK * DINNER;
    __bf16* w_xp_b = w_in_b + (size_t)DINNER * DMODEL;
    __bf16* w_out_b= w_xp_b + (size_t)NPROJ * DINNER;

    // 0. one-shot bf16 conversions (weights + input activations)
    cvt_bf16_kernel<<<(MTOK * DMODEL / 4 + 255) / 256, 256, 0, stream>>>(
        x, xb, MTOK * DMODEL / 4);
    cvt_bf16_kernel<<<(DINNER * DMODEL / 4 + 255) / 256, 256, 0, stream>>>(
        w_in, w_in_b, DINNER * DMODEL / 4);
    cvt_bf16_kernel<<<(NPROJ * DINNER / 4 + 255) / 256, 256, 0, stream>>>(
        w_xp, w_xp_b, NPROJ * DINNER / 4);
    cvt_bf16_kernel<<<(DMODEL * DINNER / 4 + 255) / 256, 256, 0, stream>>>(
        w_out, w_out_b, DMODEL * DINNER / 4);

    // 1. u = gelu(x @ in_proj_w^T + b)     M=8192 N=192 K=192; f32 + bf16 out
    gemm_bf16_wmma<DMODEL, 3><<<dim3(MTOK / 16), 128, 0, stream>>>(
        xb, w_in_b, b_in, u, ub, DINNER, DINNER, 1);

    // 2. proj = u @ x_proj_w^T             M=8192 N=72  K=192 (ld 80)
    gemm_bf16_wmma<DINNER, 2><<<dim3(MTOK / 16), 128, 0, stream>>>(
        ub, w_xp_b, nullptr, proj, (__bf16*)nullptr, NPROJ, PROJ_LD, 0);

    // 3. deltaT / deltaL (K = 12 -> VALU)
    delta_kernel<<<(MTOK * DINNER + 255) / 256, 256, 0, stream>>>(
        proj, w_dtT, b_dtT, w_dtL, b_dtL, delT, delL, MTOK * DINNER);

    // 4. 2D wavefront scan + y + gelu      384 blocks x 1024 threads, bf16 y
    scan_kernel<<<BB * DINNER, 1024, 0, stream>>>(
        u, proj, delT, delL, AT_log, AL_log, Dsk, yb);

    // 5. out = y @ out_proj_w^T + b        M=8192 N=192 K=192
    gemm_bf16_wmma<DINNER, 3><<<dim3(MTOK / 16), 128, 0, stream>>>(
        yb, w_out_b, b_out, out, (__bf16*)nullptr, DMODEL, DINNER, 0);
}
